// SelfAttention_67465346286144
// MI455X (gfx1250) — matrix operand order repaired and checked by reading, not compiled, not run on hardware
//
#include <hip/hip_runtime.h>
#include <hip/hip_bf16.h>
#include <stdint.h>

#define BATCH 8
#define CH    256
#define NTOK  4096

typedef __attribute__((ext_vector_type(16))) __bf16    v16bf;
typedef __attribute__((ext_vector_type(8)))  float     v8f;
typedef __attribute__((ext_vector_type(4)))  uint32_t  u32x4;
typedef __attribute__((ext_vector_type(4)))  float     f32x4;
typedef __attribute__((ext_vector_type(2)))  float     f32x2;

union Frag { uint32_t u[8]; u32x4 q[2]; v16bf v; };

// Hardware packed f32->bf16 convert: dst[15:0]=bf16(lo), dst[31:16]=bf16(hi).
__device__ __forceinline__ uint32_t cvt_pk_bf16(float lo, float hi) {
  uint32_t r;
  asm("v_cvt_pk_bf16_f32 %0, %1, %2" : "=v"(r) : "v"(lo), "v"(hi));
  return r;
}

__device__ __forceinline__ v8f wmma_bf16(v16bf a, v16bf b, v8f c) {
  // D = A(16x32 bf16) * B(32x16 bf16) + C(16x16 f32)
  return __builtin_amdgcn_wmma_f32_16x16x32_bf16(false, a, false, b, (short)0, c,
                                                 false, false);
}

// Load an A-style bf16 fragment (K = 2*hb + s, contiguous) -> 2x b128.
__device__ __forceinline__ void load_afrag(Frag& f, const uint16_t* base, int hb) {
  f.q[0] = *(const u32x4*)(base + 2 * hb);
  f.q[1] = *(const u32x4*)(base + 2 * hb + 8);
}
// Load a B-style bf16 fragment (K = 2v + 16*hi) -> 2x b128.
__device__ __forceinline__ void load_bfrag(Frag& f, const uint16_t* base, int hi) {
  f.q[0] = *(const u32x4*)(base + 16 * hi);
  f.q[1] = *(const u32x4*)(base + 16 * hi + 8);
}
// Load+convert an fp32 A-style fragment (weight matrix row) -> 4x b128 + 8 cvt_pk.
__device__ __forceinline__ void load_afrag_f32(Frag& f, const float* base, int hb) {
  f32x4 a = *(const f32x4*)(base + 2 * hb);
  f32x4 b = *(const f32x4*)(base + 2 * hb + 4);
  f32x4 c = *(const f32x4*)(base + 2 * hb + 8);
  f32x4 d = *(const f32x4*)(base + 2 * hb + 12);
  f.u[0] = cvt_pk_bf16(a.x, a.y); f.u[1] = cvt_pk_bf16(a.z, a.w);
  f.u[2] = cvt_pk_bf16(b.x, b.y); f.u[3] = cvt_pk_bf16(b.z, b.w);
  f.u[4] = cvt_pk_bf16(c.x, c.y); f.u[5] = cvt_pk_bf16(c.z, c.w);
  f.u[6] = cvt_pk_bf16(d.x, d.y); f.u[7] = cvt_pk_bf16(d.z, d.w);
}

// ---------------------------------------------------------------------------
// Kernel 1: qkv = W_qkv @ x + b_qkv.  Q (scaled 1/16) and K stored transposed
// [b,n,c] bf16; V stored [b,c,n] bf16.  One 16x16 tile / wave, 8 waves/block.
// ---------------------------------------------------------------------------
__global__ void qkv_kernel(const float* __restrict__ x,
                           const float* __restrict__ w,
                           const float* __restrict__ bias,
                           uint16_t* __restrict__ qs,
                           uint16_t* __restrict__ ks,
                           uint16_t* __restrict__ vs) {
  const int wave = threadIdx.x >> 5;
  const int lane = threadIdx.x & 31;
  const int hi = lane >> 4, lx = lane & 15, hb = hi * 8;

  long wid = (long)blockIdx.x * 8 + wave;          // 98304 waves total
  const int nt = (int)(wid % 256); wid /= 256;     // N tiles (4096/16)
  const int mt = (int)(wid % 48);                  // M tiles (768/16)
  const int b  = (int)(wid / 48);
  const int o0 = mt * 16, n0 = nt * 16;

  v8f acc = {0.f, 0.f, 0.f, 0.f, 0.f, 0.f, 0.f, 0.f};
  const float* xb   = x + (size_t)b * CH * NTOK;
  const float* wrow = w + (size_t)(o0 + lx) * CH;

#pragma unroll
  for (int cs = 0; cs < 8; ++cs) {
    const int c0 = cs * 32;
    Frag A, B;
    load_afrag_f32(A, wrow + c0, hb);              // A = W tile [16o x 32c]
#pragma unroll
    for (int v = 0; v < 8; ++v) {                  // B = X tile [32c x 16n]
      const float* p = xb + (size_t)(c0 + 2 * v + 16 * hi) * NTOK + n0 + lx;
      B.u[v] = cvt_pk_bf16(p[0], p[NTOK]);
    }
    acc = wmma_bf16(A.v, B.v, acc);
  }

  const int n = n0 + lx;
  if (o0 < CH) {                                   // Q tile (pre-scaled)
    uint16_t* dst = qs + (size_t)b * NTOK * CH + (size_t)n * CH + o0 + hb;
#pragma unroll
    for (int r = 0; r < 8; r += 2) {
      f32x2 bi = *(const f32x2*)(bias + o0 + r + hb);
      *(uint32_t*)(dst + r) = cvt_pk_bf16((acc[r] + bi.x) * 0.0625f,
                                          (acc[r + 1] + bi.y) * 0.0625f);
    }
  } else if (o0 < 2 * CH) {                        // K tile (transposed)
    uint16_t* dst = ks + (size_t)b * NTOK * CH + (size_t)n * CH + (o0 - CH) + hb;
#pragma unroll
    for (int r = 0; r < 8; r += 2) {
      f32x2 bi = *(const f32x2*)(bias + o0 + r + hb);
      *(uint32_t*)(dst + r) = cvt_pk_bf16(acc[r] + bi.x, acc[r + 1] + bi.y);
    }
  } else {                                         // V tile ([b,c,n], strided)
    uint16_t* dst = vs + (size_t)b * CH * NTOK +
                    (size_t)(o0 - 2 * CH + hb) * NTOK + n;
#pragma unroll
    for (int r = 0; r < 8; r += 2) {
      f32x2 bi = *(const f32x2*)(bias + o0 + r + hb);
      uint32_t pk = cvt_pk_bf16(acc[r] + bi.x, acc[r + 1] + bi.y);
      dst[(size_t)r * NTOK]       = (uint16_t)pk;
      dst[(size_t)(r + 1) * NTOK] = (uint16_t)(pk >> 16);
    }
  }
}

// ---------------------------------------------------------------------------
// Kernel 2: fused flash attention. Block = 2 waves, one 16-query tile / block.
// ---------------------------------------------------------------------------
__global__ void attn_kernel(const uint16_t* __restrict__ qs,
                            const uint16_t* __restrict__ ksr,
                            const uint16_t* __restrict__ vs,
                            uint16_t* __restrict__ os) {
  __shared__ uint16_t pLds[2][16 * 32];            // P tile [16n x 32m], per wave
  __shared__ float    aLds[2][16];                 // per-row broadcast, per wave

  const int wave = threadIdx.x >> 5;
  const int lane = threadIdx.x & 31;
  const int hi = lane >> 4, lx = lane & 15, hb = hi * 8;

  const int b  = blockIdx.x >> 8;
  const int n0 = (blockIdx.x & 255) * 16;
  const size_t nbase = (size_t)b * NTOK * CH;      // base of [b,n,c] layouts
  const int cwbase = wave * 128;                   // this wave's channel range

  // Preload Q^T fragments for all 8 k-steps (Q reused for every key tile).
  const uint16_t* qrow = qs + nbase + (size_t)(n0 + lx) * CH;
  Frag qf[8];
#pragma unroll
  for (int cs = 0; cs < 8; ++cs) load_afrag(qf[cs], qrow + cs * 32, hb);

  v8f acc[8];
  float rm[8], rl[8];
#pragma unroll
  for (int t = 0; t < 8; ++t) {
    acc[t] = (v8f){0.f, 0.f, 0.f, 0.f, 0.f, 0.f, 0.f, 0.f};
    rm[t] = -3.0e38f;
    rl[t] = 0.f;
  }

  for (int m0 = 0; m0 < NTOK; m0 += 32) {
    // ---- scores S[16n x 32m] = Q^T K -------------------------------------
    const uint16_t* krow = ksr + nbase + (size_t)(m0 + lx) * CH;
    v8f s0 = (v8f){0.f, 0.f, 0.f, 0.f, 0.f, 0.f, 0.f, 0.f};
    v8f s1 = (v8f){0.f, 0.f, 0.f, 0.f, 0.f, 0.f, 0.f, 0.f};
#pragma unroll
    for (int cs = 0; cs < 8; ++cs) {
      Frag k0, k1;
      load_bfrag(k0, krow + cs * 32, hi);
      load_bfrag(k1, krow + 16 * CH + cs * 32, hi);
      s0 = wmma_bf16(qf[cs].v, k0.v, s0);
      s1 = wmma_bf16(qf[cs].v, k1.v, s1);
    }

    // ---- online softmax over the 32 keys ---------------------------------
#pragma unroll
    for (int r = 0; r < 8; ++r) {
      float mx = fmaxf(s0[r], s1[r]);
      mx = fmaxf(mx, __shfl_xor(mx, 1));
      mx = fmaxf(mx, __shfl_xor(mx, 2));
      mx = fmaxf(mx, __shfl_xor(mx, 4));
      mx = fmaxf(mx, __shfl_xor(mx, 8));
      const float mnew = fmaxf(rm[r], mx);
      const float a = __expf(rm[r] - mnew);
      const float p0 = __expf(s0[r] - mnew);
      const float p1 = __expf(s1[r] - mnew);
      float sum = p0 + p1;
      sum += __shfl_xor(sum, 1);
      sum += __shfl_xor(sum, 2);
      sum += __shfl_xor(sum, 4);
      sum += __shfl_xor(sum, 8);
      rl[r] = rl[r] * a + sum;
      rm[r] = mnew;
      if (lx == 0) aLds[wave][r + hb] = a;         // lane0 rows0-7, lane16 rows8-15
      const uint32_t pk = cvt_pk_bf16(p0, p1);
      const int nrow = r + hb;
      pLds[wave][nrow * 32 + lx]      = (uint16_t)pk;
      pLds[wave][nrow * 32 + lx + 16] = (uint16_t)(pk >> 16);
    }
    asm volatile("" ::: "memory");                 // order LDS writes vs reads

    const float an = aLds[wave][lx];               // alpha for this O column n
    Frag pf;                                       // B = P^T [32m x 16n]
    pf.q[0] = *(const u32x4*)&pLds[wave][lx * 32 + 16 * hi];
    pf.q[1] = *(const u32x4*)&pLds[wave][lx * 32 + 16 * hi + 8];

    // ---- O[c,n] = alpha*O + V x P^T --------------------------------------
    const uint16_t* vbase = vs + (size_t)b * CH * NTOK +
                            (size_t)(cwbase + lx) * NTOK + m0;
#pragma unroll
    for (int ct = 0; ct < 8; ++ct) {
      Frag vf;                                     // A = V tile [16c x 32m]
      load_afrag(vf, vbase + (size_t)ct * 16 * NTOK, hb);
      v8f cin = acc[ct] * an;
      acc[ct] = wmma_bf16(vf.v, pf.v, cin);
    }
  }

  // ---- finalize: O /= l, store bf16 [b,c,n] ------------------------------
#pragma unroll
  for (int r = 0; r < 8; ++r) {
    if (lx == 0) aLds[wave][r + hb] = 1.0f / rl[r];
  }
  asm volatile("" ::: "memory");
  const float ln = aLds[wave][lx];
  uint16_t* obase = os + (size_t)b * CH * NTOK +
                    (size_t)(cwbase + hb) * NTOK + n0 + lx;
#pragma unroll
  for (int ct = 0; ct < 8; ++ct) {
#pragma unroll
    for (int r = 0; r < 8; r += 2) {
      uint32_t pk = cvt_pk_bf16(acc[ct][r] * ln, acc[ct][r + 1] * ln);
      uint16_t* p = obase + (size_t)(ct * 16 + r) * NTOK;
      p[0]    = (uint16_t)pk;
      p[NTOK] = (uint16_t)(pk >> 16);
    }
  }
}

// ---------------------------------------------------------------------------
// Kernel 3: out = W_proj @ O + b_proj + x  (fp32 result, residual add).
// ---------------------------------------------------------------------------
__global__ void proj_kernel(const float* __restrict__ x,
                            const float* __restrict__ wp,
                            const float* __restrict__ bp,
                            const uint16_t* __restrict__ os,
                            float* __restrict__ out) {
  const int wave = threadIdx.x >> 5;
  const int lane = threadIdx.x & 31;
  const int hi = lane >> 4, lx = lane & 15, hb = hi * 8;

  long wid = (long)blockIdx.x * 8 + wave;          // 32768 waves total
  const int nt = (int)(wid % 256); wid /= 256;
  const int mt = (int)(wid % 16);
  const int b  = (int)(wid / 16);
  const int o0 = mt * 16, n0 = nt * 16;

  v8f acc = {0.f, 0.f, 0.f, 0.f, 0.f, 0.f, 0.f, 0.f};
  const uint16_t* ob = os + (size_t)b * CH * NTOK + n0 + lx;
  const float* wrow = wp + (size_t)(o0 + lx) * CH;

#pragma unroll
  for (int cs = 0; cs < 8; ++cs) {
    const int c0 = cs * 32;
    Frag A, B;
    load_afrag_f32(A, wrow + c0, hb);              // A = W_proj tile
#pragma unroll
    for (int v = 0; v < 8; ++v) {                  // B = O tile (already bf16)
      const uint16_t* p = ob + (size_t)(c0 + 2 * v + 16 * hi) * NTOK;
      B.u[v] = (uint32_t)p[0] | ((uint32_t)p[NTOK] << 16);
    }
    acc = wmma_bf16(A.v, B.v, acc);
  }

#pragma unroll
  for (int r = 0; r < 8; ++r) {
    const int o = o0 + r + hb;
    const size_t idx = (size_t)b * CH * NTOK + (size_t)o * NTOK + n0 + lx;
    out[idx] = acc[r] + bp[o] + x[idx];
  }
}

// ---------------------------------------------------------------------------
extern "C" void kernel_launch(void* const* d_in, const int* in_sizes, int n_in,
                              void* d_out, int out_size, void* d_ws, size_t ws_size,
                              hipStream_t stream) {
  const float* x     = (const float*)d_in[0];
  const float* w_qkv = (const float*)d_in[1];
  const float* b_qkv = (const float*)d_in[2];
  const float* w_prj = (const float*)d_in[3];
  const float* b_prj = (const float*)d_in[4];
  float* out = (float*)d_out;

  const size_t elems = (size_t)BATCH * CH * NTOK;  // 8,388,608
  uint16_t* qs = (uint16_t*)d_ws;                  // [b,n,c] bf16, 16 MB
  uint16_t* ks = qs + elems;                       // [b,n,c] bf16, 16 MB
  uint16_t* vs = ks + elems;                       // [b,c,n] bf16, 16 MB
  uint16_t* os = vs + elems;                       // [b,c,n] bf16, 16 MB

  // 1) QKV projection: 8 * 48 * 256 wave-tiles, 8 waves / block.
  qkv_kernel<<<12288, 256, 0, stream>>>(x, w_qkv, b_qkv, qs, ks, vs);

  // 2) Fused flash attention: one block (2 waves) per 16-query tile.
  attn_kernel<<<BATCH * (NTOK / 16), 64, 0, stream>>>(qs, ks, vs, os);

  // 3) Output projection + bias + residual: 8 * 16 * 256 wave-tiles.
  proj_kernel<<<4096, 256, 0, stream>>>(x, w_prj, b_prj, os, out);
}
